// KANLinear_68539088110032
// MI455X (gfx1250) — compile-verified, matrix operand order checked by reading, not run on hardware
//
#include <hip/hip_runtime.h>

// ---------------------------------------------------------------------------
// KANLinear fused forward for MI455X (gfx1250, wave32, WMMA + TDM).
//
//   out[b,o] = sum_i silu(x[b,i]) * Wb[o,i]
//            + sum_{i,k} basis_k(x[b,i]) * Ws[i,k,o] * Sc[i,o]
//
// One f16 GEMM, expanded K = 1024 (base) + 8192 (spline) = 9216, f32 accum
// via v_wmma_f32_16x16x32_f16.  155 GFLOP vs ~104 MB compulsory HBM traffic
// -> compute bound -> tensor cores.
//
// PREP path (workspace >= 162 MB) feeds LDS with TENSOR_LOAD_TO_LDS (Tensor
// Data Mover): one 2-D tile descriptor per fill, zero VGPR staging,
// TENSORcnt + s_wait_tensorcnt 0 before each K-step barrier.  Spline
// weights are prepped pre-scaled AND transposed (swT[o][r]) so all four
// fills are plain row-major strided tiles; LDS row padding (32->40 halves)
// comes from the descriptor pad fields (16 DWORDs data + 4 DWORDs pad).
// Fused fallback path (tiny ws): division-free Cox-de Boor + v_rcp silu.
//
// This toolchain's __builtin_amdgcn_tensor_load_to_lds takes 6 args
// (g0 v4u, g1 v8i, g2 v4i, g3 v4i, extra v8i, cpol i32) per the compile
// probe -- groups 2/3 and the extra group are zero for 2-D tiles.
// ---------------------------------------------------------------------------

typedef _Float16 v16h __attribute__((ext_vector_type(16)));
typedef _Float16 v8h  __attribute__((ext_vector_type(8)));
typedef _Float16 v4h  __attribute__((ext_vector_type(4)));
typedef float    v8f  __attribute__((ext_vector_type(8)));
typedef unsigned int v4u __attribute__((ext_vector_type(4)));
typedef int      v8i __attribute__((ext_vector_type(8)));
typedef int      v4i __attribute__((ext_vector_type(4)));

#define BATCH 8192
#define IN_F  1024
#define OUT_F 1024
#define KEXP  (IN_F * 8)      // expanded spline K

#define BM 128
#define BN 256
#define KT 32
#define NT 512                // threads per block = 16 wave32s
#define LDSW 40               // LDS row stride in halves (80 B, 16B aligned)
#define SA_STRIDE (BM * LDSW) // halves per A buffer (10 KB)
#define SB_STRIDE (BN * LDSW) // halves per B buffer (20 KB)

#if __has_builtin(__builtin_amdgcn_tensor_load_to_lds) && \
    __has_builtin(__builtin_amdgcn_s_wait_tensorcnt)
#define KAN_TDM 1
#else
#define KAN_TDM 0
#endif

union Frag16 { v16h v; v8h h[2]; };

__device__ __forceinline__ float silu_f(float v) {
    return v * __builtin_amdgcn_rcpf(1.0f + __expf(-v));   // v_exp + v_rcp
}

// Division-free Cox-de Boor on the uniform knot grid t_j = -1 + (j-3)*0.4.
// u = (x - t_0)/h = fma(x, 2.5, 5.5);  d_j = u - j = (x - t_j)/h.
// Level k:  b_k[j] = (d_j*b[j] - d_{j+k+1}*b[j+1]) * (1/k)   (denoms = k*h).
__device__ __forceinline__ void bspline8(float xv, float* __restrict__ o8) {
    const float u  = __builtin_fmaf(xv, 2.5f, 5.5f);
    const float fl = floorf(u);
    float d[12];
#pragma unroll
    for (int j = 0; j < 12; ++j) d[j] = u - (float)j;
    float b0[11];
#pragma unroll
    for (int j = 0; j < 11; ++j) b0[j] = (fl == (float)j) ? 1.0f : 0.0f;
    float b1[10];
#pragma unroll
    for (int j = 0; j < 10; ++j) b1[j] = d[j] * b0[j] - d[j + 2] * b0[j + 1];
    float b2[9];
#pragma unroll
    for (int j = 0; j < 9; ++j)
        b2[j] = (d[j] * b1[j] - d[j + 3] * b1[j + 1]) * 0.5f;
#pragma unroll
    for (int j = 0; j < 8; ++j)
        o8[j] = (d[j] * b2[j] - d[j + 4] * b2[j + 1]) * (1.0f / 3.0f);
}

#if KAN_TDM
// One TDM 2-D tile DMA: `rows` x 32 halves from global (row stride
// `gstride` halves) into LDS at byte offset lds_byte, rows padded 32->40
// halves via pad fields (interval 16 DWORDs, amount 4 DWORDs).
// D# layout per CDNA5 ISA.
__device__ __forceinline__ void tdm_tile_f16(const _Float16* gsrc,
                                             unsigned lds_byte,
                                             int rows, int gstride_halves) {
    unsigned long long ga = (unsigned long long)(size_t)gsrc;
    v4u g0;
    g0[0] = 1u;                                        // count=1 (valid D#)
    g0[1] = lds_byte;                                  // lds_addr (bytes)
    g0[2] = (unsigned)ga;                              // global_addr[31:0]
    g0[3] = (unsigned)((ga >> 32) & 0x1FFFFFFull)      // global_addr[56:32]
          | (2u << 30);                                // type=2 ("image")
    v8i g1;
    g1[0] = (int)((1u << 16)                           // data_size = 2 bytes
                | (1u << 20)                           // pad_enable
                | (3u << 22)                           // pad_interval: 16 DW
                | (3u << 25));                         // pad_amount:   4 DW
    g1[1] = (int)((32u & 0xFFFFu) << 16);              // tensor_dim0 = 32
    g1[2] = (int)(((unsigned)rows & 0xFFFFu) << 16);   // tensor_dim1 = rows
    g1[3] = (int)(32u << 16);                          // tile_dim0 = 32
    g1[4] = (int)((unsigned)rows & 0xFFFFu);           // tile_dim1 = rows
    g1[5] = gstride_halves;                            // dim0_stride lo32
    g1[6] = 0;                                         // dim0_stride hi
    g1[7] = 0;                                         // dim1_stride (2-D)
    v4i z4; z4[0] = z4[1] = z4[2] = z4[3] = 0;         // groups 2/3 unused
    v8i z8;
#pragma unroll
    for (int j = 0; j < 8; ++j) z8[j] = 0;             // extra group unused
    __builtin_amdgcn_tensor_load_to_lds(g0, g1, z4, z4, z8, 0);
}
#endif

// ------------------------------ prep kernels -------------------------------

__global__ __launch_bounds__(256) void kan_prep_x(
    const float* __restrict__ x, _Float16* __restrict__ xs16) {
    size_t idx = (size_t)blockIdx.x * 256 + threadIdx.x;   // float4 index
    float4 v = ((const float4*)x)[idx];
    v4h h;
    h[0] = (_Float16)silu_f(v.x); h[1] = (_Float16)silu_f(v.y);
    h[2] = (_Float16)silu_f(v.z); h[3] = (_Float16)silu_f(v.w);
    ((v4h*)xs16)[idx] = h;
}

__global__ __launch_bounds__(256) void kan_prep_wb(
    const float* __restrict__ bw, _Float16* __restrict__ bw16) {
    size_t idx = (size_t)blockIdx.x * 256 + threadIdx.x;   // float4 index
    float4 v = ((const float4*)bw)[idx];
    v4h h;
    h[0] = (_Float16)v.x; h[1] = (_Float16)v.y;
    h[2] = (_Float16)v.z; h[3] = (_Float16)v.w;
    ((v4h*)bw16)[idx] = h;
}

// Pre-scale spline weights by the scaler and TRANSPOSE: swT[o][r], r=i*8+k.
// 32x32 tile transpose through LDS; reads coalesced along o, writes
// coalesced along r.
__global__ __launch_bounds__(256) void kan_prep_swT(
    const float* __restrict__ sw, const float* __restrict__ sc,
    _Float16* __restrict__ swT) {
    __shared__ float tile[32][33];
    const int t  = threadIdx.x;
    const int rb = blockIdx.y * 32;            // expanded-K tile base
    const int ob = blockIdx.x * 32;            // out-feature tile base
    const int c  = t & 31, w = t >> 5;         // 8 rows per pass
#pragma unroll
    for (int it = 0; it < 4; ++it) {
        int r = rb + w + it * 8;
        int o = ob + c;
        tile[w + it * 8][c] =
            sw[(size_t)r * OUT_F + o] * sc[(size_t)(r >> 3) * OUT_F + o];
    }
    __syncthreads();
#pragma unroll
    for (int it = 0; it < 4; ++it) {
        int o = ob + w + it * 8;
        int r = rb + c;
        swT[(size_t)o * KEXP + r] = (_Float16)tile[c][w + it * 8];
    }
}

__global__ __launch_bounds__(256) void kan_prep_basis(
    const float* __restrict__ x, _Float16* __restrict__ bas16) {
    size_t idx = (size_t)blockIdx.x * 256 + threadIdx.x;   // (b, i) pair
    float bs[8];
    bspline8(x[idx], bs);
    v8h h;
#pragma unroll
    for (int j = 0; j < 8; ++j) h[j] = (_Float16)bs[j];
    ((v8h*)bas16)[idx] = h;
}

// ------------------------------- GEMM kernel -------------------------------

template <bool PREP>
__global__ __launch_bounds__(NT) void kan_gemm(
    const float* __restrict__ x,
    const float* __restrict__ base_w,
    const float* __restrict__ spline_w,
    const float* __restrict__ scaler,
    const _Float16* __restrict__ xs16,
    const _Float16* __restrict__ bw16,
    const _Float16* __restrict__ swT,     // [OUT_F][KEXP] pre-scaled, transposed
    const _Float16* __restrict__ bas16,
    float* __restrict__ out)
{
    __shared__ _Float16 sA[2 * SA_STRIDE];   // 20 KB, double buffered
    __shared__ _Float16 sB[2 * SB_STRIDE];   // 40 KB, double buffered

    const int tid  = threadIdx.x;
    const int lane = tid & 31;
    const int wave = tid >> 5;               // 16 waves
    const int wm   = wave >> 3;              // 0..1 -> 64 rows
    const int wn   = wave & 7;               // 0..7 -> 32 cols
    const int m0   = blockIdx.y * BM;
    const int n0   = blockIdx.x * BN;

    const int hsel = lane >> 4;
    const int l15  = lane & 15;

    v8f acc[4][2];                           // 64 VGPRs of accumulators
#pragma unroll
    for (int f = 0; f < 4; ++f)
#pragma unroll
        for (int g = 0; g < 2; ++g)
#pragma unroll
            for (int e = 0; e < 8; ++e) acc[f][g][e] = 0.0f;

    // 16x16 fragment from LDS tile (row-major [row][K], stride LDSW):
    // lanes 0-15 -> K 0-7 / 16-23 ; lanes 16-31 -> K 8-15 / 24-31.
    auto ldFrag = [&](const _Float16* base, int rowBase) -> v16h {
        const _Float16* p = base + (rowBase + l15) * LDSW + hsel * 8;
        Frag16 u;
        u.h[0] = *(const v8h*)(p);
        u.h[1] = *(const v8h*)(p + 16);
        return u.v;
    };

    auto mmaStep = [&](int buf) {
        const _Float16* A = sA + buf * SA_STRIDE;
        const _Float16* B = sB + buf * SB_STRIDE;
        v16h bf[2];
#pragma unroll
        for (int g = 0; g < 2; ++g) bf[g] = ldFrag(B, wn * 32 + g * 16);
#pragma unroll
        for (int f = 0; f < 4; ++f) {
            v16h af = ldFrag(A, wm * 64 + f * 16);
#pragma unroll
            for (int g = 0; g < 2; ++g)
                acc[f][g] = __builtin_amdgcn_wmma_f32_16x16x32_f16(
                    false, af, false, bf[g], (short)0, acc[f][g],
                    false, false);
        }
    };

    // fill-stage index precomputes (512 threads)
    const int rowA = tid >> 2;               // 0..127  (A tiles: 8 halves/thr)
    const int cbA  = (tid & 3) * 8;          // 0/8/16/24
    const int rowB = tid >> 1;               // 0..255  (B tiles: 16/thr)
    const int cbB  = (tid & 1) * 16;         // 0/16
    const int blA2 = tid >> 2;               // fused basis: 1 (b,i)/thread
    const int ilA2 = tid & 3;
    const int kk   = tid & 31;               // fused phase-2 B: K column
    const int nb   = (tid >> 5) * 16;        // fused phase-2 B: 16 N rows

#if KAN_TDM
    const unsigned ldsA0 = (unsigned)(size_t)(const void*)sA;
    const unsigned ldsB0 = (unsigned)(size_t)(const void*)sB;
#endif

    // ------------- phase 1 fill: A=silu(x) tile, B=base_w^T tile -----------
    auto fillP1 = [&](int buf, int k0) {
        if constexpr (PREP) {
#if KAN_TDM
            if (wave == 0) {
                tdm_tile_f16(xs16 + (size_t)m0 * IN_F + k0,
                             ldsA0 + buf * SA_STRIDE * 2, BM, IN_F);
                tdm_tile_f16(bw16 + (size_t)n0 * IN_F + k0,
                             ldsB0 + buf * SB_STRIDE * 2, BN, IN_F);
            }
#else
            _Float16* ap = sA + buf * SA_STRIDE + rowA * LDSW + cbA;
            *(v8h*)ap = *(const v8h*)(xs16 + (size_t)(m0 + rowA) * IN_F + k0 + cbA);
            _Float16* bp = sB + buf * SB_STRIDE + rowB * LDSW + cbB;
            const v8h* src = (const v8h*)(bw16 + (size_t)(n0 + rowB) * IN_F + k0 + cbB);
            ((v8h*)bp)[0] = src[0];
            ((v8h*)bp)[1] = src[1];
#endif
        } else {
            _Float16* ap = sA + buf * SA_STRIDE + rowA * LDSW + cbA;
            const float4* xp = (const float4*)(x + (size_t)(m0 + rowA) * IN_F + k0 + cbA);
            float4 a = xp[0], b = xp[1];
            v8h h;
            h[0] = (_Float16)silu_f(a.x); h[1] = (_Float16)silu_f(a.y);
            h[2] = (_Float16)silu_f(a.z); h[3] = (_Float16)silu_f(a.w);
            h[4] = (_Float16)silu_f(b.x); h[5] = (_Float16)silu_f(b.y);
            h[6] = (_Float16)silu_f(b.z); h[7] = (_Float16)silu_f(b.w);
            *(v8h*)ap = h;
            _Float16* bp = sB + buf * SB_STRIDE + rowB * LDSW + cbB;
            const float4* wp = (const float4*)(base_w + (size_t)(n0 + rowB) * IN_F + k0 + cbB);
#pragma unroll
            for (int q = 0; q < 2; ++q) {
                float4 a2 = wp[2 * q], b2 = wp[2 * q + 1];
                v8h h2;
                h2[0] = (_Float16)a2.x; h2[1] = (_Float16)a2.y;
                h2[2] = (_Float16)a2.z; h2[3] = (_Float16)a2.w;
                h2[4] = (_Float16)b2.x; h2[5] = (_Float16)b2.y;
                h2[6] = (_Float16)b2.z; h2[7] = (_Float16)b2.w;
                ((v8h*)bp)[q] = h2;
            }
        }
    };

    // --------- phase 2 fill: A=basis tile, B=scaled spline weights ---------
    auto fillP2 = [&](int buf, int i0) {
        if constexpr (PREP) {
#if KAN_TDM
            if (wave == 0) {
                tdm_tile_f16(bas16 + (size_t)m0 * KEXP + i0 * 8,
                             ldsA0 + buf * SA_STRIDE * 2, BM, KEXP);
                tdm_tile_f16(swT + (size_t)n0 * KEXP + i0 * 8,
                             ldsB0 + buf * SB_STRIDE * 2, BN, KEXP);
            }
#else
            _Float16* ap = sA + buf * SA_STRIDE + rowA * LDSW + cbA;
            *(v8h*)ap = *(const v8h*)(bas16 + (size_t)(m0 + rowA) * KEXP + i0 * 8 + cbA);
            _Float16* bp = sB + buf * SB_STRIDE + rowB * LDSW + cbB;
            const v8h* src = (const v8h*)(swT + (size_t)(n0 + rowB) * KEXP + i0 * 8 + cbB);
            ((v8h*)bp)[0] = src[0];
            ((v8h*)bp)[1] = src[1];
#endif
        } else {
            float bs[8];
            bspline8(x[(size_t)(m0 + blA2) * IN_F + i0 + ilA2], bs);
            v8h h;
#pragma unroll
            for (int j = 0; j < 8; ++j) h[j] = (_Float16)bs[j];
            *(v8h*)(sA + buf * SA_STRIDE + blA2 * LDSW + ilA2 * 8) = h;
            const int r = i0 * 8 + kk;                   // expanded K row
            const int i = r >> 3;
            _Float16* bbase = sB + buf * SB_STRIDE;
            const float4* sw4 = (const float4*)(spline_w + (size_t)r * OUT_F + n0 + nb);
            const float4* sc4 = (const float4*)(scaler   + (size_t)i * OUT_F + n0 + nb);
#pragma unroll
            for (int q = 0; q < 4; ++q) {
                float4 w = sw4[q], s = sc4[q];
                bbase[(nb + q * 4 + 0) * LDSW + kk] = (_Float16)(w.x * s.x);
                bbase[(nb + q * 4 + 1) * LDSW + kk] = (_Float16)(w.y * s.y);
                bbase[(nb + q * 4 + 2) * LDSW + kk] = (_Float16)(w.z * s.z);
                bbase[(nb + q * 4 + 3) * LDSW + kk] = (_Float16)(w.w * s.w);
            }
        }
    };

    auto tensorWait = [&]() {
#if KAN_TDM
        if constexpr (PREP) __builtin_amdgcn_s_wait_tensorcnt(0);
#endif
    };

    // --------------------------- main K loops ------------------------------
    fillP1(0, 0);
    constexpr int S1 = IN_F / KT;                       // 32 steps
    for (int s = 0; s < S1; ++s) {
        tensorWait();
        __syncthreads();
        if (s + 1 < S1) fillP1((s + 1) & 1, (s + 1) * KT);
        mmaStep(s & 1);
    }

    constexpr int S2 = IN_F / 4;                        // 256 steps
    fillP2(0, 0);                                       // buf0 free: last read s=S1-2
    for (int s = 0; s < S2; ++s) {
        tensorWait();
        __syncthreads();
        if (s + 1 < S2) fillP2((s + 1) & 1, (s + 1) * 4);
        mmaStep(s & 1);
    }

    // ------------------------------ epilogue -------------------------------
    // C/D: VGPR r -> row r (lanes 0-15) / 8+r (lanes 16-31), col = lane&15.
#pragma unroll
    for (int f = 0; f < 4; ++f) {
        const int rowD = m0 + wm * 64 + f * 16 + hsel * 8;
#pragma unroll
        for (int g = 0; g < 2; ++g) {
            const int col = n0 + wn * 32 + g * 16 + l15;
            float* op = out + (size_t)rowD * OUT_F + col;
#pragma unroll
            for (int r = 0; r < 8; ++r)
                op[(size_t)r * OUT_F] = acc[f][g][r];
        }
    }
}

// -------------------------------- launcher ---------------------------------

extern "C" void kernel_launch(void* const* d_in, const int* in_sizes, int n_in,
                              void* d_out, int out_size, void* d_ws, size_t ws_size,
                              hipStream_t stream) {
    (void)in_sizes; (void)n_in; (void)out_size;
    const float* x        = (const float*)d_in[0];
    // d_in[1] = grid : unused, uniform knots are compile-time constants
    const float* base_w   = (const float*)d_in[2];
    const float* spline_w = (const float*)d_in[3];
    const float* scaler   = (const float*)d_in[4];
    float* out = (float*)d_out;

    const size_t nXs  = (size_t)BATCH * IN_F;        //  8,388,608 halves
    const size_t nBw  = (size_t)OUT_F * IN_F;        //  1,048,576
    const size_t nSw  = (size_t)KEXP * OUT_F;        //  8,388,608
    const size_t nBas = (size_t)BATCH * KEXP;        // 67,108,864
    const size_t need = (nXs + nBw + nSw + nBas) * sizeof(_Float16); // ~162 MB

    dim3 grid(OUT_F / BN, BATCH / BM);               // (4, 64) = 256 blocks

    if (ws_size >= need) {
        _Float16* p     = (_Float16*)d_ws;
        _Float16* xs16  = p;  p += nXs;
        _Float16* bw16  = p;  p += nBw;
        _Float16* swT   = p;  p += nSw;
        _Float16* bas16 = p;
        kan_prep_x    <<<(unsigned)(nXs / 4 / 256), 256, 0, stream>>>(x, xs16);
        kan_prep_wb   <<<(unsigned)(nBw / 4 / 256), 256, 0, stream>>>(base_w, bw16);
        kan_prep_swT  <<<dim3(OUT_F / 32, KEXP / 32), 256, 0, stream>>>(spline_w, scaler, swT);
        kan_prep_basis<<<(unsigned)(nXs / 256),     256, 0, stream>>>(x, bas16);
        kan_gemm<true><<<grid, dim3(NT), 0, stream>>>(
            x, base_w, spline_w, scaler, xs16, bw16, swT, bas16, out);
    } else {
        kan_gemm<false><<<grid, dim3(NT), 0, stream>>>(
            x, base_w, spline_w, scaler, nullptr, nullptr, nullptr, nullptr, out);
    }
}
